// MultiHeadAttention_21595095564454
// MI455X (gfx1250) — matrix operand order repaired and checked by reading, not compiled, not run on hardware
//
#include <hip/hip_runtime.h>
#include <hip/hip_bf16.h>

typedef _Float16 half_t;
typedef __attribute__((ext_vector_type(4)))  _Float16 v4h;
typedef __attribute__((ext_vector_type(8)))  _Float16 v8h;
typedef __attribute__((ext_vector_type(16))) _Float16 v16h;
typedef __attribute__((ext_vector_type(8)))  float    v8f;
typedef __attribute__((ext_vector_type(4)))  int      i32x4;

// ---------------------------------------------------------------------------
// Async global -> LDS copy (CDNA5 GLOBAL_LOAD_ASYNC_TO_LDS_B128, ASYNCcnt),
// with a synchronous fallback if the builtin is unavailable.
// Probe-learned signature: (global int4*, shared int4*, imm offset, imm cpol).
// ---------------------------------------------------------------------------
#if __has_builtin(__builtin_amdgcn_global_load_async_to_lds_b128)
#define HAS_ASYNC_LDS 1
__device__ __forceinline__ void async_copy16(const half_t* g, half_t* l) {
  __builtin_amdgcn_global_load_async_to_lds_b128(
      (__attribute__((address_space(1))) i32x4*)g,
      (__attribute__((address_space(3))) i32x4*)l, 0, 0);
}
#else
#define HAS_ASYNC_LDS 0
__device__ __forceinline__ void async_copy16(const half_t* g, half_t* l) {
  *(v8h*)l = *(const v8h*)g;
}
#endif

__device__ __forceinline__ void wait_async() {
#if HAS_ASYNC_LDS
#if __has_builtin(__builtin_amdgcn_s_wait_asynccnt)
  __builtin_amdgcn_s_wait_asynccnt(0);
#else
  asm volatile("s_wait_asynccnt 0x0" ::: "memory");
#endif
#endif
}

// ---------------------------------------------------------------------------
// WMMA helpers (CDNA5 f16 inputs, f32 accumulate)
// ---------------------------------------------------------------------------
__device__ __forceinline__ v8f wmma_f16(v16h a, v16h b, v8f c) {
  // (neg_a, A, neg_b, B, c_mod, C, reuse_a, reuse_b)
  return __builtin_amdgcn_wmma_f32_16x16x32_f16(false, a, false, b, (short)0, c,
                                                false, false);
}

__device__ __forceinline__ v16h make_v16h(v8h lo, v8h hi) {
  v16h r;
#pragma unroll
  for (int i = 0; i < 8; ++i) { r[i] = lo[i]; r[8 + i] = hi[i]; }
  return r;
}

// A fragment: 16x32 f16, tile stored row-major [row][k], stride ld halves.
// Lane l holds row (l&15); element i -> K=(l>>4)*8+i, element 8+i -> K=16+(l>>4)*8+i.
__device__ __forceinline__ v16h load_frag_a(const half_t* base, int ld, int lane) {
  const int row = lane & 15;
  const int kb  = (lane >> 4) * 8;
  const half_t* p = base + row * ld + kb;
  return make_v16h(*(const v8h*)p, *(const v8h*)(p + 16));
}

// B fragment: 32x16 f16 (KxN), tile stored in LDS as [n][k], stride ld.
// Lane l holds column (l&15); element i -> K=(l>>4)*8+i, element 8+i -> K=16+(l>>4)*8+i.
__device__ __forceinline__ v16h load_frag_b(const half_t* base, int ld, int lane) {
  const int col = lane & 15;
  const int kb  = (lane >> 4) * 8;
  const half_t* p = base + col * ld + kb;
  return make_v16h(*(const v8h*)p, *(const v8h*)(p + 16));
}

// ---------------------------------------------------------------------------
// fp32 -> f16 conversion (vectorized x4)
// ---------------------------------------------------------------------------
__global__ __launch_bounds__(256) void cvt_f32_f16(const float* __restrict__ s,
                                                   half_t* __restrict__ d, int n4) {
  int i = blockIdx.x * 256 + threadIdx.x;
  if (i < n4) {
    float4 f = ((const float4*)s)[i];
    v4h o;
    o[0] = (half_t)f.x; o[1] = (half_t)f.y; o[2] = (half_t)f.z; o[3] = (half_t)f.w;
    *(v4h*)(d + (size_t)i * 4) = o;
  }
}

// ---------------------------------------------------------------------------
// fp32 [K,N] -> f16 transposed [N,K], tiled through LDS (coalesced both sides)
// Grid (N/32, K/32), block (32,8).
// ---------------------------------------------------------------------------
__global__ __launch_bounds__(256) void cvt_transpose_f32_f16(
    const float* __restrict__ src, half_t* __restrict__ dst, int K, int N) {
  __shared__ float tile[32][33];
  const int k0 = blockIdx.y * 32;
  const int n0 = blockIdx.x * 32;
#pragma unroll
  for (int i = threadIdx.y; i < 32; i += 8)
    tile[i][threadIdx.x] = src[(size_t)(k0 + i) * N + n0 + threadIdx.x];
  __syncthreads();
#pragma unroll
  for (int i = threadIdx.y; i < 32; i += 8)
    dst[(size_t)(n0 + i) * K + k0 + threadIdx.x] = (half_t)tile[threadIdx.x][i];
}

// ---------------------------------------------------------------------------
// GEMM: C[M,N] = A[M,K] (f16, row-major) * Bt[N,K]^T (f16, pre-transposed)
//       + bias[N]
// Block tile 128x256, BK=64, 256 threads = 8 waves (2 x 4), wave tile 64x64.
// M % 128 == 0, N % 256 == 0, K % 64 == 0 (true here).
// ---------------------------------------------------------------------------
template <bool OUT_F16>
__global__ __launch_bounds__(256) void gemm_f16_bias(
    const half_t* __restrict__ A, const half_t* __restrict__ Bt,
    const float* __restrict__ bias, void* __restrict__ outp,
    int M, int Nn, int K) {
  __shared__ half_t As[128 * 72];   // [row][k], k padded to 72
  __shared__ half_t Bs[256 * 72];   // [col][k], k padded to 72

  const int tid   = threadIdx.x;
  const int lane  = tid & 31;
  const int wave  = tid >> 5;
  const int waveM = wave >> 2;   // 0..1
  const int waveN = wave & 3;    // 0..3
  const int bM = blockIdx.y * 128;
  const int bN = blockIdx.x * 256;

  v8f zero = {};
  v8f acc[4][4];
#pragma unroll
  for (int sm = 0; sm < 4; ++sm)
#pragma unroll
    for (int sn = 0; sn < 4; ++sn) acc[sm][sn] = zero;

  const int ldr = tid >> 3;          // 0..31 row group base
  const int ldk = (tid & 7) << 3;    // 0..56

  for (int kt = 0; kt < K; kt += 64) {
    __syncthreads();   // previous tile fully consumed
    // ---- A tile: 128 rows x 64 k = 1024 b128 chunks (4 per thread)
#pragma unroll
    for (int t = 0; t < 4; ++t) {
      int r = ldr + t * 32;
      async_copy16(A + (size_t)(bM + r) * K + kt + ldk, &As[r * 72 + ldk]);
    }
    // ---- B tile: 256 cols x 64 k = 2048 b128 chunks (8 per thread)
#pragma unroll
    for (int t = 0; t < 8; ++t) {
      int n = ldr + t * 32;
      async_copy16(Bt + (size_t)(bN + n) * K + kt + ldk, &Bs[n * 72 + ldk]);
    }
    // ---- prefetch next k-tile into caches while we compute this one
    if (kt + 64 < K) {
      __builtin_prefetch(A + (size_t)(bM + ldr) * K + kt + 64 + ldk, 0, 1);
      __builtin_prefetch(Bt + (size_t)(bN + ldr) * K + kt + 64 + ldk, 0, 1);
      __builtin_prefetch(Bt + (size_t)(bN + 128 + ldr) * K + kt + 64 + ldk, 0, 1);
    }
    wait_async();
    __syncthreads();
    // ---- compute: 2 k-steps of 32, 16 WMMAs each
#pragma unroll
    for (int ks = 0; ks < 2; ++ks) {
      const int k0 = ks * 32;
      v16h af[4], bf[4];
#pragma unroll
      for (int sm = 0; sm < 4; ++sm)
        af[sm] = load_frag_a(&As[(waveM * 64 + sm * 16) * 72 + k0], 72, lane);
#pragma unroll
      for (int sn = 0; sn < 4; ++sn)
        bf[sn] = load_frag_b(&Bs[(waveN * 64 + sn * 16) * 72 + k0], 72, lane);
#pragma unroll
      for (int sm = 0; sm < 4; ++sm)
#pragma unroll
        for (int sn = 0; sn < 4; ++sn)
          acc[sm][sn] = wmma_f16(af[sm], bf[sn], acc[sm][sn]);
    }
  }

  // ---- epilogue: C layout row = j + 8*(lane>>4), col = lane&15
  const int colLocal = lane & 15;
  const int rowHalf  = (lane >> 4) * 8;
#pragma unroll
  for (int sn = 0; sn < 4; ++sn) {
    const int col = bN + waveN * 64 + sn * 16 + colLocal;
    const float bv = bias[col];
#pragma unroll
    for (int sm = 0; sm < 4; ++sm) {
      const int rowBase = bM + waveM * 64 + sm * 16 + rowHalf;
#pragma unroll
      for (int j = 0; j < 8; ++j) {
        float v = acc[sm][sn][j] + bv;
        if (OUT_F16)
          ((half_t*)outp)[(size_t)(rowBase + j) * Nn + col] = (half_t)v;
        else
          ((float*)outp)[(size_t)(rowBase + j) * Nn + col] = v;
      }
    }
  }
}

// ---------------------------------------------------------------------------
// Fused flash attention. qkv f16 [B,N,3,H,D]. out f16 [B,N,H*D].
// Grid (N/128, H, B). 256 threads = 8 waves; each wave: 16 q-rows.
// ---------------------------------------------------------------------------
__global__ __launch_bounds__(256) void attn_fwd(const half_t* __restrict__ qkv,
                                                half_t* __restrict__ outh) {
  constexpr int Nq = 2048;
  constexpr float SCALE = 0.125f;   // 1/sqrt(64)

  __shared__ half_t Kt[64 * 72];      // [key][d]  (already B-layout for QK^T)
  __shared__ half_t Vts[64 * 72];     // [d][key]  (transposed V, B-layout for PV)
  __shared__ half_t Ps[8 * 16 * 72];  // per-wave P tile [row][key]

  const int tid  = threadIdx.x;
  const int lane = tid & 31;
  const int wave = tid >> 5;
  const int b  = blockIdx.z;
  const int h  = blockIdx.y;
  const int qt = blockIdx.x;
  const int q0 = qt * 128 + wave * 16;

  const int arow = lane & 15;
  const int kb   = (lane >> 4) * 8;

  // Q fragments held in registers (A-matrix of QK^T, K-dim = head dim = 64)
  const half_t* qp =
      qkv + ((size_t)((b * Nq + q0 + arow) * 3 + 0)) * 1024 + h * 64;
  const v16h qf0 = make_v16h(*(const v8h*)(qp + kb), *(const v8h*)(qp + 16 + kb));
  const v16h qf1 = make_v16h(*(const v8h*)(qp + 32 + kb), *(const v8h*)(qp + 48 + kb));

  float mrow[8], lrow[8];
  v8f zero = {};
  v8f o[4];
#pragma unroll
  for (int j = 0; j < 8; ++j) { mrow[j] = -3.0e38f; lrow[j] = 0.0f; }
#pragma unroll
  for (int ds = 0; ds < 4; ++ds) o[ds] = zero;

  for (int kt0 = 0; kt0 < Nq; kt0 += 64) {
    __syncthreads();   // previous tile's Kt/Vts fully consumed
    // ---- cooperative load: K via async DMA, V transposed via VGPR scatter
#pragma unroll
    for (int t = 0; t < 2; ++t) {
      int c   = tid + t * 256;
      int key = c >> 3;
      int dd  = (c & 7) << 3;
      const half_t* kp =
          qkv + ((size_t)((b * Nq + kt0 + key) * 3 + 1)) * 1024 + h * 64 + dd;
      async_copy16(kp, &Kt[key * 72 + dd]);
      const half_t* vp =
          qkv + ((size_t)((b * Nq + kt0 + key) * 3 + 2)) * 1024 + h * 64 + dd;
      v8h vv = *(const v8h*)vp;
#pragma unroll
      for (int i = 0; i < 8; ++i) Vts[(dd + i) * 72 + key] = vv[i];
    }
    // prefetch next K/V tile
    if (kt0 + 64 < Nq) {
      int key = tid >> 3, dd = (tid & 7) << 3;
      __builtin_prefetch(
          qkv + ((size_t)((b * Nq + kt0 + 64 + key) * 3 + 1)) * 1024 + h * 64 + dd,
          0, 1);
      __builtin_prefetch(
          qkv + ((size_t)((b * Nq + kt0 + 64 + key) * 3 + 2)) * 1024 + h * 64 + dd,
          0, 1);
    }
    wait_async();
    __syncthreads();

    // ---- S = Q K^T  (4 key-subtiles x 2 chained WMMAs over d=64)
    v8f s[4];
#pragma unroll
    for (int ks = 0; ks < 4; ++ks) {
      s[ks] = zero;
      v16h b0 = load_frag_b(&Kt[(ks * 16) * 72 + 0], 72, lane);
      v16h b1 = load_frag_b(&Kt[(ks * 16) * 72 + 32], 72, lane);
      s[ks] = wmma_f16(qf0, b0, s[ks]);
      s[ks] = wmma_f16(qf1, b1, s[ks]);
    }
#pragma unroll
    for (int ks = 0; ks < 4; ++ks)
#pragma unroll
      for (int j = 0; j < 8; ++j) s[ks][j] *= SCALE;

    // ---- online softmax (row = j + 8*(lane>>4); row spans 16 lanes)
    float mx[8];
#pragma unroll
    for (int j = 0; j < 8; ++j)
      mx[j] = fmaxf(fmaxf(s[0][j], s[1][j]), fmaxf(s[2][j], s[3][j]));
#pragma unroll
    for (int off = 8; off >= 1; off >>= 1)
#pragma unroll
      for (int j = 0; j < 8; ++j)
        mx[j] = fmaxf(mx[j], __shfl_xor(mx[j], off, 32));

#pragma unroll
    for (int j = 0; j < 8; ++j) {
      float mn = fmaxf(mrow[j], mx[j]);
      float al = __expf(mrow[j] - mn);
      mrow[j] = mn;
      lrow[j] *= al;
#pragma unroll
      for (int ds = 0; ds < 4; ++ds) o[ds][j] *= al;
    }

    float rs[8];
#pragma unroll
    for (int j = 0; j < 8; ++j) rs[j] = 0.0f;
#pragma unroll
    for (int ks = 0; ks < 4; ++ks)
#pragma unroll
      for (int j = 0; j < 8; ++j) {
        float p = __expf(s[ks][j] - mrow[j]);
        s[ks][j] = p;
        rs[j] += p;
      }
#pragma unroll
    for (int off = 8; off >= 1; off >>= 1)
#pragma unroll
      for (int j = 0; j < 8; ++j) rs[j] += __shfl_xor(rs[j], off, 32);
#pragma unroll
    for (int j = 0; j < 8; ++j) lrow[j] += rs[j];

    // ---- P (C-layout) -> LDS -> A-layout for PV
    half_t* pw = &Ps[wave * 16 * 72];
#pragma unroll
    for (int ks = 0; ks < 4; ++ks)
#pragma unroll
      for (int j = 0; j < 8; ++j)
        pw[(j + ((lane >> 4) << 3)) * 72 + ks * 16 + (lane & 15)] =
            (half_t)s[ks][j];
    __syncthreads();

    // ---- O += P V  (4 d-subtiles x 2 k-steps over 64 keys)
#pragma unroll
    for (int ds = 0; ds < 4; ++ds) {
#pragma unroll
      for (int kstep = 0; kstep < 2; ++kstep) {
        v16h pa = load_frag_a(&Ps[wave * 16 * 72 + kstep * 32], 72, lane);
        v16h vb = load_frag_b(&Vts[(ds * 16) * 72 + kstep * 32], 72, lane);
        o[ds] = wmma_f16(pa, vb, o[ds]);
      }
    }
  }

  // ---- normalize + store f16 [B,N,H*D]
#pragma unroll
  for (int ds = 0; ds < 4; ++ds)
#pragma unroll
    for (int j = 0; j < 8; ++j) {
      int qrow = q0 + j + ((lane >> 4) << 3);
      int col  = h * 64 + ds * 16 + (lane & 15);
      outh[(size_t)(b * Nq + qrow) * 1024 + col] = (half_t)(o[ds][j] / lrow[j]);
    }
}

// ---------------------------------------------------------------------------
// Launcher
// ---------------------------------------------------------------------------
extern "C" void kernel_launch(void* const* d_in, const int* in_sizes, int n_in,
                              void* d_out, int out_size, void* d_ws, size_t ws_size,
                              hipStream_t stream) {
  const float* x     = (const float*)d_in[0];   // [4,2048,1024]
  const float* Wqkv  = (const float*)d_in[1];   // [1024,3072]
  const float* bqkv  = (const float*)d_in[2];   // [3072]
  const float* Wproj = (const float*)d_in[3];   // [1024,1024]
  const float* bproj = (const float*)d_in[4];   // [1024]
  float* out = (float*)d_out;                   // [4,2048,1024] fp32

  const size_t nX = 8388608, nWq = 3145728, nWp = 1048576;
  const size_t nQKV = 25165824, nAttn = 8388608;

  half_t* ws    = (half_t*)d_ws;
  half_t* xh    = ws;                 // [8192,1024]        f16
  half_t* wqt   = xh + nX;            // [3072,1024] (W^T)  f16
  half_t* wpt   = wqt + nWq;          // [1024,1024] (W^T)  f16
  half_t* qkvh  = wpt + nWp;          // [B,N,3,H,D]        f16
  half_t* atth  = qkvh + nQKV;        // [B,N,C]            f16
  (void)nAttn; (void)ws_size; (void)in_sizes; (void)n_in; (void)out_size;

  // 1) fp32 -> f16 conversions; weights transposed to [N,K]
  cvt_f32_f16<<<(int)(nX / 4 / 256), 256, 0, stream>>>(x, xh, (int)(nX / 4));
  cvt_transpose_f32_f16<<<dim3(3072 / 32, 1024 / 32), dim3(32, 8), 0, stream>>>(
      Wqkv, wqt, 1024, 3072);
  cvt_transpose_f32_f16<<<dim3(1024 / 32, 1024 / 32), dim3(32, 8), 0, stream>>>(
      Wproj, wpt, 1024, 1024);

  // 2) QKV GEMM: [8192,1024] x [1024,3072] + bias -> f16 [B,N,3,H,D]
  gemm_f16_bias<true><<<dim3(3072 / 256, 8192 / 128), 256, 0, stream>>>(
      xh, wqt, bqkv, qkvh, 8192, 3072, 1024);

  // 3) fused attention -> f16 [B,N,C]
  attn_fwd<<<dim3(2048 / 128, 16, 4), 256, 0, stream>>>(qkvh, atth);

  // 4) projection GEMM: [8192,1024] x [1024,1024] + bias -> fp32 out
  gemm_f16_bias<false><<<dim3(1024 / 256, 8192 / 128), 256, 0, stream>>>(
      atth, wpt, bproj, out, 8192, 1024, 1024);
}
